// MotifSelectionPool_26388279066708
// MI455X (gfx1250) — compile-verified, matrix operand order checked by reading, not run on hardware
//
#include <hip/hip_runtime.h>
#include <hip/hip_bf16.h>
#include <math.h>

// CDNA5 / gfx1250: wave32, WMMA 16x16x32 bf16 (f32 accumulate), async LDS loads
typedef __bf16 bf16_t;
typedef bf16_t   v16bf __attribute__((ext_vector_type(16)));
typedef float    v8f   __attribute__((ext_vector_type(8)));
typedef unsigned v8u   __attribute__((ext_vector_type(8)));

#define TILE_K 32

// ---------------- Stage A: column sums of motif (+1 identity) -> dinv --------
__global__ void k_colsum_partial(const float* __restrict__ motif,
                                 float* __restrict__ deg, int N, int rows_per) {
    int j = blockIdx.x * blockDim.x + threadIdx.x;
    if (j >= N) return;
    int r0 = blockIdx.y * rows_per;
    int r1 = min(N, r0 + rows_per);
    float s = 0.f;
    for (int i = r0; i < r1; ++i) s += motif[(size_t)i * N + j];
    atomicAdd(&deg[j], s);
}

__global__ void k_dinv(const float* __restrict__ deg, float* __restrict__ dinv, int N) {
    int j = blockIdx.x * blockDim.x + threadIdx.x;
    if (j >= N) return;
    float d = deg[j] + 1.0f;             // + identity diagonal
    dinv[j] = (d > 0.f) ? rsqrtf(d) : 0.f;
}

// ------- Stage B: Yt[d,i] = dinv[i] * (x @ W^T)[i,d]  (bf16, TRANSPOSED) -----
__global__ void k_xw_y(const float* __restrict__ x, const float* __restrict__ W,
                       const float* __restrict__ dinv, bf16_t* __restrict__ Yt,
                       int N, int D) {
    __shared__ float xs[128];
    int i = blockIdx.x;
    int d = threadIdx.x;                 // blockDim.x == D == 128
    xs[d] = x[(size_t)i * D + d];
    __syncthreads();
    const float* wr = W + (size_t)d * D;
    float acc = 0.f;
    #pragma unroll 8
    for (int k = 0; k < 128; ++k) acc += xs[k] * wr[k];
    Yt[(size_t)d * N + i] = (bf16_t)(dinv[i] * acc);   // k-contiguous for async
}

// ---------------- Stage C: h = tanh(dinv[j]*(motif^T @ Y + Y[j]) + b) --------
// One workgroup (8 wave32) per 16-row output strip; wave w owns d-tile w.
// B tile streamed via global_load_async_to_lds_b64 (ASYNCcnt), A tile converted
// f32->bf16 through VGPRs. Guard-free main K loop + one guarded tail step.
__global__ __launch_bounds__(256)
void k_gcn_gemm(const float* __restrict__ motif, const bf16_t* __restrict__ Yt,
                const float* __restrict__ dinv, const float* __restrict__ bgcn,
                float* __restrict__ h, int N) {
    __shared__ bf16_t Ash[16][34];       // [j-col m][k]; 17-bank stride, pairs 4B-aligned
    __shared__ bf16_t Bsh[128][36];      // [d][k]; 72B stride: 8B-aligned, conflict-free

    const int t      = threadIdx.x;
    const int lane   = t & 31;
    const int wave   = t >> 5;           // 0..7 -> d tile
    const int m      = lane & 15;
    const int half   = lane >> 4;
    const int j_base = blockIdx.x * 16;  // N % 16 == 0 here
    const int d_base = wave * 16;

    v8f acc = {};

    // VGPR g of a 16-bit fragment holds K pair (K0, K0+1):
    int K0[8];
    #pragma unroll
    for (int g = 0; g < 8; ++g) K0[g] = ((g < 4) ? 0 : 16) + 8 * half + 2 * (g & 3);

    const int ak = t >> 3;               // A staging row 0..31
    const int ac = (t & 7) * 2;          // A staging col 0,2,...,14
    const float* ap = motif + (size_t)ak * N + j_base + ac;   // loop-carried
    const size_t astep = (size_t)TILE_K * N;

    const int kmain = N & ~(TILE_K - 1); // guard-free region

    for (int kc = 0; kc < kmain; kc += TILE_K) {
        // ---- B tile: Bsh[d][0..31] <= Yt[d][kc..kc+31], async global->LDS ----
        #pragma unroll
        for (int rr = 0; rr < 4; ++rr) {
            int idx = t + 256 * rr;              // 0..1023 -> 128 rows x 8 segs
            int d   = idx >> 3;                  // 0..127
            int s   = (idx & 7) * 4;             // k offset, 8B granules
            unsigned lds  = (unsigned)(uintptr_t)&Bsh[d][s];
            const bf16_t* src = Yt + (size_t)d * N + kc + s;
            asm volatile("global_load_async_to_lds_b64 %0, %1, off"
                         :: "v"(lds), "v"(src) : "memory");
        }
        // ---- A tile: f32 -> bf16 through VGPRs, pointer-incremented ----------
        {
            float a0 = ap[0], a1 = ap[1];
            __builtin_prefetch(ap + astep, 0, 1);            // speculative: safe OOB
            Ash[ac][ak]     = (bf16_t)a0;
            Ash[ac + 1][ak] = (bf16_t)a1;
            ap += astep;
        }
        asm volatile("s_wait_asynccnt 0x0" ::: "memory");    // async B visible
        __syncthreads();

        v8u au, bu;
        #pragma unroll
        for (int g = 0; g < 8; ++g) {
            au[g] = *(const unsigned*)&Ash[m][K0[g]];            // A[m][K..K+1]
            bu[g] = *(const unsigned*)&Bsh[d_base + m][K0[g]];   // B[K..K+1][n=m]
        }
        v16bf a = __builtin_bit_cast(v16bf, au);
        v16bf b = __builtin_bit_cast(v16bf, bu);
        acc = __builtin_amdgcn_wmma_f32_16x16x32_bf16(
                  false, a, false, b, (short)0, acc, false, false);
        __syncthreads();
    }

    if (kmain < N) {                     // single guarded tail K-step
        #pragma unroll
        for (int rr = 0; rr < 4; ++rr) {
            int idx = t + 256 * rr;
            int d   = idx >> 3;
            int s   = (idx & 7) * 4;
            if (kmain + s + 4 <= N) {
                unsigned lds  = (unsigned)(uintptr_t)&Bsh[d][s];
                const bf16_t* src = Yt + (size_t)d * N + kmain + s;
                asm volatile("global_load_async_to_lds_b64 %0, %1, off"
                             :: "v"(lds), "v"(src) : "memory");
            } else {
                uint2 z = {0u, 0u};
                *(uint2*)&Bsh[d][s] = z;
            }
        }
        {
            float a0 = 0.f, a1 = 0.f;
            if (kmain + ak < N) { a0 = ap[0]; a1 = ap[1]; }
            Ash[ac][ak]     = (bf16_t)a0;
            Ash[ac + 1][ak] = (bf16_t)a1;
        }
        asm volatile("s_wait_asynccnt 0x0" ::: "memory");
        __syncthreads();

        v8u au, bu;
        #pragma unroll
        for (int g = 0; g < 8; ++g) {
            au[g] = *(const unsigned*)&Ash[m][K0[g]];
            bu[g] = *(const unsigned*)&Bsh[d_base + m][K0[g]];
        }
        v16bf a = __builtin_bit_cast(v16bf, au);
        v16bf b = __builtin_bit_cast(v16bf, bu);
        acc = __builtin_amdgcn_wmma_f32_16x16x32_bf16(
                  false, a, false, b, (short)0, acc, false, false);
        __syncthreads();
    }

    // C/D layout: VGPR v, lane L -> row M = v + 8*(L>>4), col = L&15
    #pragma unroll
    for (int v = 0; v < 8; ++v) {
        int j = j_base + v + 8 * half;
        int d = d_base + m;
        float yjd = (float)Yt[(size_t)d * N + j];            // identity term
        float val = dinv[j] * (acc[v] + yjd) + bgcn[d];
        h[(size_t)j * 128 + d] = tanhf(val);
    }
}

// ---------------- Stage D: scores = h @ w_score + b  (one wave32 per row) ----
__global__ void k_scores(const float* __restrict__ h, const float* __restrict__ w,
                         const float* __restrict__ bsc, float* __restrict__ scores,
                         float* __restrict__ out_scores, int N, int D) {
    int wave = threadIdx.x >> 5;
    int lane = threadIdx.x & 31;
    int j = blockIdx.x * (blockDim.x >> 5) + wave;
    if (j >= N) return;
    float s = 0.f;
    for (int d = lane; d < D; d += 32) s += h[(size_t)j * D + d] * w[d];
    for (int off = 16; off > 0; off >>= 1) s += __shfl_xor(s, off, 32);
    if (lane == 0) { float v = s + bsc[0]; scores[j] = v; out_scores[j] = v; }
}

// ---------------- Stage E: exact top-k membership via O(N^2) rank ------------
__global__ void k_rank(const float* __restrict__ scores, int* __restrict__ keep,
                       int N, int K) {
    __shared__ float sb[256];
    int j = blockIdx.x * 256 + threadIdx.x;
    float sj = (j < N) ? scores[j] : 0.f;
    int cnt = 0;
    for (int t0 = 0; t0 < N; t0 += 256) {
        int ti = t0 + threadIdx.x;
        sb[threadIdx.x] = (ti < N) ? scores[ti] : -__builtin_inff();
        __syncthreads();
        int lim = min(256, N - t0);
        for (int u = 0; u < lim; ++u) {
            float s = sb[u];
            cnt += ((s > sj) || (s == sj && (t0 + u) < j)) ? 1 : 0;
        }
        __syncthreads();
    }
    if (j < N) keep[j] = (cnt < K) ? 1 : 0;
}

// ---------------- Stage F: single-block scan -> sorted indices ---------------
__global__ void k_scan_idx(const int* __restrict__ keep, int* __restrict__ idx,
                           float* __restrict__ out_topidx, int N) {
    __shared__ int buf[1024];
    __shared__ int base;
    if (threadIdx.x == 0) base = 0;
    __syncthreads();
    for (int c0 = 0; c0 < N; c0 += 1024) {
        int j = c0 + threadIdx.x;
        int v = (j < N) ? keep[j] : 0;
        buf[threadIdx.x] = v;
        __syncthreads();
        for (int off = 1; off < 1024; off <<= 1) {          // Hillis-Steele
            int add = (threadIdx.x >= off) ? buf[threadIdx.x - off] : 0;
            __syncthreads();
            buf[threadIdx.x] += add;
            __syncthreads();
        }
        int pos = base + buf[threadIdx.x] - v;              // exclusive
        if (j < N && v) { idx[pos] = j; out_topidx[pos] = (float)j; }
        __syncthreads();
        if (threadIdx.x == 0) base += buf[1023];
        __syncthreads();
    }
}

// ---------------- Stage G: gathers -------------------------------------------
__global__ void k_gather_x(const float* __restrict__ h, const int* __restrict__ idx,
                           float* __restrict__ xpool, int K, int D) {
    int e = blockIdx.x * blockDim.x + threadIdx.x;
    if (e >= K * D) return;
    int r = e / D, d = e - r * D;
    xpool[e] = h[(size_t)idx[r] * D + d];
}

__global__ void k_gather_pools(const float* __restrict__ adj,
                               const float* __restrict__ motif,
                               const int* __restrict__ idx,
                               float* __restrict__ adj_pool,
                               float* __restrict__ motif_pool, int N, int K) {
    int r = blockIdx.x;
    int src = idx[r];
    const float* arow = adj   + (size_t)src * N;
    const float* mrow = motif + (size_t)src * N;
    for (int c = threadIdx.x; c < K; c += blockDim.x) {
        int col = idx[c];                    // sorted -> good line locality
        adj_pool  [(size_t)r * K + c] = arow[col];
        motif_pool[(size_t)r * K + c] = mrow[col];
    }
}

// ---------------- Host launcher ----------------------------------------------
extern "C" void kernel_launch(void* const* d_in, const int* in_sizes, int n_in,
                              void* d_out, int out_size, void* d_ws, size_t ws_size,
                              hipStream_t stream) {
    const float* x      = (const float*)d_in[0];
    const float* adj    = (const float*)d_in[1];
    const float* motif  = (const float*)d_in[2];
    const float* W      = (const float*)d_in[3];
    const float* bgcn   = (const float*)d_in[4];
    const float* wscore = (const float*)d_in[5];
    const float* bscore = (const float*)d_in[6];

    const int D = in_sizes[4];               // b_gcn length = 128
    const int N = in_sizes[0] / D;           // 10000
    const int K = (N + 1) / 2;               // ceil(0.5*N) = 5000

    // workspace carve-up (256B aligned)
    char* ws = (char*)d_ws; size_t off = 0;
    auto carve = [&](size_t bytes) -> void* {
        void* p = ws + off; off = (off + bytes + 255) & ~(size_t)255; return p;
    };
    float*  deg    = (float*) carve((size_t)N * 4);
    float*  dinv   = (float*) carve((size_t)N * 4);
    bf16_t* Yt     = (bf16_t*)carve((size_t)N * D * 2);   // [D][N] transposed
    float*  h      = (float*) carve((size_t)N * D * 4);
    float*  scores = (float*) carve((size_t)N * 4);
    int*    keep   = (int*)   carve((size_t)N * 4);
    int*    idx    = (int*)   carve((size_t)K * 4);

    float* out        = (float*)d_out;       // tuple concat, return order
    float* out_xpool  = out;
    float* out_adj    = out_xpool + (size_t)K * D;
    float* out_motif  = out_adj   + (size_t)K * K;
    float* out_scores = out_motif + (size_t)K * K;
    float* out_topidx = out_scores + N;

    hipMemsetAsync(deg, 0, (size_t)N * 4, stream);

    {   // column sums: 32 row-chunks for parallelism, atomic combine
        int rows_per = (N + 31) / 32;
        dim3 g((N + 255) / 256, 32);
        k_colsum_partial<<<g, 256, 0, stream>>>(motif, deg, N, rows_per);
    }
    k_dinv<<<(N + 255) / 256, 256, 0, stream>>>(deg, dinv, N);
    k_xw_y<<<N, D, 0, stream>>>(x, W, dinv, Yt, N, D);
    k_gcn_gemm<<<(N + 15) / 16, 256, 0, stream>>>(motif, Yt, dinv, bgcn, h, N);
    k_scores<<<(N + 7) / 8, 256, 0, stream>>>(h, wscore, bscore, scores, out_scores, N, D);
    k_rank<<<(N + 255) / 256, 256, 0, stream>>>(scores, keep, N, K);
    k_scan_idx<<<1, 1024, 0, stream>>>(keep, idx, out_topidx, N);
    k_gather_x<<<((size_t)K * D + 255) / 256, 256, 0, stream>>>(h, idx, out_xpool, K, D);
    k_gather_pools<<<K, 256, 0, stream>>>(adj, motif, idx, out_adj, out_motif, N, K);
}
